// GNN_drug_33019708572239
// MI455X (gfx1250) — compile-verified, matrix operand order checked by reading, not run on hardware
//
#include <hip/hip_runtime.h>
#include <hip/hip_bf16.h>

typedef _Float16 half_t;
typedef __attribute__((ext_vector_type(4)))  _Float16 v4h;
typedef __attribute__((ext_vector_type(8)))  _Float16 v8h;
typedef __attribute__((ext_vector_type(16))) _Float16 v16h;
typedef __attribute__((ext_vector_type(8)))  float    v8f;

#define D128 128
#define F_IN 77
#define BN_EPS 1e-5f

// ---------------------------------------------------------------- utilities
__global__ void fill_zero4(float4* __restrict__ p, long n4) {
    long i = (long)blockIdx.x * blockDim.x + threadIdx.x;
    if (i < n4) p[i] = make_float4(0.f, 0.f, 0.f, 0.f);
}

// pad x [N,77] -> A [N,128] (cols 77..127 zero)
__global__ void pad_x(const float* __restrict__ x, float* __restrict__ A, int N) {
    long i = (long)blockIdx.x * blockDim.x + threadIdx.x;
    if (i >= (long)N * D128) return;
    int c = (int)(i & (D128 - 1));
    long row = i >> 7;
    A[i] = (c < F_IN) ? x[row * F_IN + c] : 0.f;
}

// ------------------------------------------------- edge scatter (one wave / edge)
__global__ void edge_agg(const float* __restrict__ A, const int* __restrict__ ei,
                         float* __restrict__ B, int E) {
    long t = (long)blockIdx.x * blockDim.x + threadIdx.x;
    int e = (int)(t >> 5);
    int lane = (int)(t & 31);
    if (e >= E) return;
    int src = ei[e];
    int dst = ei[(long)E + e];
    float4 v = ((const float4*)(A + (long)src * D128))[lane];
    float* dp = B + (long)dst * D128 + lane * 4;
    atomicAdd(dp + 0, v.x);
    atomicAdd(dp + 1, v.y);
    atomicAdd(dp + 2, v.z);
    atomicAdd(dp + 3, v.w);
}

// Ah = (half)(A + B), 4 elements per thread
__global__ void hsum_half(const float* __restrict__ A, const float* __restrict__ B,
                          half_t* __restrict__ Ah, long n4) {
    long i = (long)blockIdx.x * blockDim.x + threadIdx.x;
    if (i >= n4) return;
    float4 a = ((const float4*)A)[i];
    float4 b = ((const float4*)B)[i];
    v4h h;
    h.x = (_Float16)(a.x + b.x);
    h.y = (_Float16)(a.y + b.y);
    h.z = (_Float16)(a.z + b.z);
    h.w = (_Float16)(a.w + b.w);
    ((v4h*)Ah)[i] = h;
}

// ---------------------------------- pre-swizzle weights into WMMA B-fragment order
// frag index t = ((kt*8 + nt)*32 + lane); each t owns 16 halves (32B).
// Lane layout mirrors the ISA 16-bit operand layout: lane&15 = tile column,
// (lane>>4)*8 = K-half base; halves 0..7 = K {khb..khb+7}, 8..15 = K {16+khb..}.
__global__ void prep_w(const float* __restrict__ W, int din, half_t* __restrict__ frag) {
    int t = blockIdx.x * blockDim.x + threadIdx.x;
    if (t >= 1024) return;
    int lane = t & 31;
    int nt = (t >> 5) & 7;
    int kt = t >> 8;
    int c = lane & 15;
    int khb = (lane >> 4) * 8;
    int col = nt * 16 + c;
    half_t* o = frag + (long)t * 16;
#pragma unroll
    for (int j = 0; j < 8; ++j) {
        int k0 = kt * 32 + khb + j;
        int k1 = kt * 32 + 16 + khb + j;
        o[j]     = (k0 < din) ? (half_t)W[(long)k0 * D128 + col] : (half_t)0.f;
        o[8 + j] = (k1 < din) ? (half_t)W[(long)k1 * D128 + col] : (half_t)0.f;
    }
}

// ------------------------------------------------------------- WMMA GEMM
// C[N,128] = relu(Ah[N,128] @ Wfrag + bias).  One wave -> 16 rows x 128 cols.
// 8 waves / block -> 128 rows / block.  Requires N % 16 == 0 (true: 100000).
template <bool STORE_HALF>
__global__ void gemm_wmma(const half_t* __restrict__ Ah,
                          const half_t* __restrict__ Wfrag,
                          const float* __restrict__ bias,
                          void* __restrict__ out, int nrows) {
    int wave = threadIdx.x >> 5;
    int lane = threadIdx.x & 31;
    int row_base = (blockIdx.x * 8 + wave) * 16;
    if (row_base >= nrows) return;      // wave-uniform: EXEC stays all-ones

    int r = lane & 15;                  // A row within tile
    int khb = (lane >> 4) * 8;          // K-half base per ISA A layout
    const half_t* arow = Ah + (long)(row_base + r) * D128;

    v8f acc[8] = {};
#pragma unroll
    for (int kt = 0; kt < 4; ++kt) {
        v8h alo = *(const v8h*)(arow + kt * 32 + khb);
        v8h ahi = *(const v8h*)(arow + kt * 32 + 16 + khb);
        v16h a = __builtin_shufflevector(alo, ahi, 0, 1, 2, 3, 4, 5, 6, 7,
                                         8, 9, 10, 11, 12, 13, 14, 15);
#pragma unroll
        for (int nt = 0; nt < 8; ++nt) {
            v16h b = *(const v16h*)(Wfrag + (long)(((kt * 8 + nt) * 32 + lane)) * 16);
            acc[nt] = __builtin_amdgcn_wmma_f32_16x16x32_f16(
                false, a, false, b, (short)0, acc[nt], false, false);
        }
    }

    int rowoff = (lane >> 4) * 8;       // C layout: lanes 16..31 hold M = j+8
#pragma unroll
    for (int nt = 0; nt < 8; ++nt) {
        int col = nt * 16 + (lane & 15);
        float bv = bias[col];
#pragma unroll
        for (int j = 0; j < 8; ++j) {
            float v = acc[nt][j] + bv;
            v = v > 0.f ? v : 0.f;
            long idx = (long)(row_base + rowoff + j) * D128 + col;
            if (STORE_HALF) ((half_t*)out)[idx] = (half_t)v;
            else            ((float*)out)[idx]  = v;
        }
    }
}

// ------------------------------------------------------------- BatchNorm
__global__ void bn_stats(const float* __restrict__ U, float* __restrict__ stats, int N) {
    int c = threadIdx.x;                // 128 threads, one column each
    float s = 0.f, sq = 0.f;
    for (int r = blockIdx.x; r < N; r += gridDim.x) {
        float v = U[(long)r * D128 + c];
        s += v; sq += v * v;
    }
    atomicAdd(&stats[c], s);
    atomicAdd(&stats[D128 + c], sq);
}

__global__ void bn_apply(const float* __restrict__ U, const float* __restrict__ stats,
                         const float* __restrict__ gamma, const float* __restrict__ beta,
                         float* __restrict__ A, long nelem, float invN) {
    long i = (long)blockIdx.x * blockDim.x + threadIdx.x;
    if (i >= nelem) return;
    int c = (int)(i & (D128 - 1));
    float mu = stats[c] * invN;
    float var = stats[D128 + c] * invN - mu * mu;
    float v = (U[i] - mu) * rsqrtf(var + BN_EPS) * gamma[c] + beta[c];
    A[i] = v;
}

// ------------------------------------------------------------- segment max
__device__ __forceinline__ int lower_bound_i(const int* a, int n, int key) {
    int lo = 0, hi = n;
    while (lo < hi) { int mid = (lo + hi) >> 1; if (a[mid] < key) lo = mid + 1; else hi = mid; }
    return lo;
}

__global__ void seg_max(const float* __restrict__ A, const int* __restrict__ batch,
                        float* __restrict__ out, int N, int layer) {
    int g = blockIdx.x;
    int lo = lower_bound_i(batch, N, g);
    int hi = lower_bound_i(batch, N, g + 1);
    int c = threadIdx.x;                // 128 threads, one column each
    float m = -INFINITY;
    for (int n = lo; n < hi; ++n) m = fmaxf(m, A[(long)n * D128 + c]);
    out[(long)g * 384 + layer * D128 + c] = m;
}

// ------------------------------------------------------------- host driver
extern "C" void kernel_launch(void* const* d_in, const int* in_sizes, int n_in,
                              void* d_out, int out_size, void* d_ws, size_t ws_size,
                              hipStream_t stream) {
    const float* x   = (const float*)d_in[0];
    const int*   ei  = (const int*)d_in[1];
    const int*   bat = (const int*)d_in[2];
    const int N = in_sizes[2];
    const int E = in_sizes[1] / 2;
    const int G = out_size / 384;
    const long nelem = (long)N * D128;
    const long n4 = nelem / 4;

    char* ws = (char*)d_ws;
    float*  A     = (float*)ws;                    ws += nelem * 4;   // activations / BN out
    float*  B     = (float*)ws;                    ws += nelem * 4;   // agg buffer
    float*  U     = (float*)ws;                    ws += nelem * 4;   // MLP f32 out
    half_t* Ah    = (half_t*)ws;                   ws += nelem * 2;   // f16 GEMM1 input
    half_t* Th    = (half_t*)ws;                   ws += nelem * 2;   // f16 GEMM2 input
    half_t* W1f   = (half_t*)ws;                   ws += 16384 * 2;   // swizzled weights
    half_t* W2f   = (half_t*)ws;                   ws += 16384 * 2;
    float*  stats = (float*)ws;                    ws += 1024;

    const int TB = 256;
    const int blk_e4   = (int)((nelem + TB - 1) / TB);      // elementwise, 1/thread
    const int blk_v4   = (int)((n4 + TB - 1) / TB);         // elementwise, 4/thread
    const int blk_edge = (int)(((long)E * 32 + TB - 1) / TB);
    const int blk_gemm = (N / 16 + 7) / 8;                  // N % 16 == 0
    const float invN = 1.0f / (float)N;
    float* out = (float*)d_out;

    // layer 0 input: zero-padded x
    pad_x<<<blk_e4, TB, 0, stream>>>(x, A, N);

    for (int layer = 0; layer < 3; ++layer) {
        const float* w1 = (const float*)d_in[3 + 6 * layer + 0];
        const float* b1 = (const float*)d_in[3 + 6 * layer + 1];
        const float* w2 = (const float*)d_in[3 + 6 * layer + 2];
        const float* b2 = (const float*)d_in[3 + 6 * layer + 3];
        const float* gm = (const float*)d_in[3 + 6 * layer + 4];
        const float* be = (const float*)d_in[3 + 6 * layer + 5];
        const int din = (layer == 0) ? F_IN : D128;

        // 1) agg = scatter_add(A[src] -> dst)
        fill_zero4<<<blk_v4, TB, 0, stream>>>((float4*)B, n4);
        edge_agg<<<blk_edge, TB, 0, stream>>>(A, ei, B, E);

        // 2) Ah = f16(A + agg)
        hsum_half<<<blk_v4, TB, 0, stream>>>(A, B, Ah, n4);

        // 3) swizzle weights into WMMA fragment order
        prep_w<<<4, TB, 0, stream>>>(w1, din, W1f);
        prep_w<<<4, TB, 0, stream>>>(w2, D128, W2f);

        // 4) MLP: relu(h@w1+b1) -> Th (f16);  relu(Th@w2+b2) -> U (f32)
        gemm_wmma<true ><<<blk_gemm, TB, 0, stream>>>(Ah, W1f, b1, (void*)Th, N);
        gemm_wmma<false><<<blk_gemm, TB, 0, stream>>>(Th, W2f, b2, (void*)U, N);

        // 5) BatchNorm (training stats, biased var); writes next layer input A
        fill_zero4<<<1, 64, 0, stream>>>((float4*)stats, 64);
        bn_stats<<<512, 128, 0, stream>>>(U, stats, N);
        bn_apply<<<blk_e4, TB, 0, stream>>>(U, stats, gm, be, A, nelem, invN);

        // 6) JK-cat + segment_max: write this layer's 128-col slice of [G,384]
        seg_max<<<G, 128, 0, stream>>>(A, bat, out, N, layer);
    }
}
